// Gemma4VisionAttention_46969762349453
// MI455X (gfx1250) — matrix operand order repaired and checked by reading, not compiled, not run on hardware
//
#include <hip/hip_runtime.h>
#include <math.h>
#include <stdint.h>

#define HIDDEN 1536
#define SEQ 2048
#define BATCH 2
#define NQ 12
#define NKV 4
#define HD 128

typedef __attribute__((ext_vector_type(16))) __bf16 v16bf;
typedef __attribute__((ext_vector_type(8)))  __bf16 v8bf;
typedef __attribute__((ext_vector_type(8)))  float  v8f;
typedef __attribute__((ext_vector_type(4)))  float  v4f;
typedef __attribute__((ext_vector_type(4)))  unsigned int v4u;
typedef __attribute__((ext_vector_type(4)))  int v4i;
typedef __attribute__((ext_vector_type(8)))  int v8i;

union U16 { v16bf v; v8bf h[2]; };

static __device__ __forceinline__ v8f wmma_bf16(v16bf a, v16bf b, v8f c) {
  return __builtin_amdgcn_wmma_f32_16x16x32_bf16(false, a, false, b, (short)0, c, false, false);
}

static __device__ __forceinline__ unsigned pack_bf16(float a, float b) {
  unsigned short ua = __builtin_bit_cast(unsigned short, (__bf16)a);
  unsigned short ub = __builtin_bit_cast(unsigned short, (__bf16)b);
  return (unsigned)ua | ((unsigned)ub << 16);
}

// A fragment 16(M)x32(K): lane<16 -> elems 0..7 = K0..7, 8..15 = K16..23
//                         lane>=16 -> elems 0..7 = K8..15, 8..15 = K24..31
static __device__ __forceinline__ v16bf ldsA(const __bf16* base, int stride, int lane) {
  const __bf16* p = base + (lane & 15) * stride + ((lane >> 4) << 3);
  U16 u;
  u.h[0] = *(const v8bf*)(p);
  u.h[1] = *(const v8bf*)(p + 16);
  return u.v;
}

// B fragment 32(K)x16(N): N = lane&15; lane<16 -> elems 0..7 = K0..7, 8..15 = K16..23,
// lane>=16 -> elems 0..7 = K8..15, 8..15 = K24..31 (strided reads).
static __device__ __forceinline__ v16bf ldsB(const __bf16* base, int stride, int lane) {
  const __bf16* p = base + (lane & 15) + ((lane >> 4) << 3) * stride;
  v16bf r;
  #pragma unroll
  for (int e = 0; e < 8; ++e) { r[e] = p[e * stride]; r[e + 8] = p[(e + 16) * stride]; }
  return r;
}

// Issue a TDM load of a 2-D bf16 tile (tile_d x tile_keys) from global (row
// stride HD elems) into LDS with 16B padding after every 256B (V_STRIDE=136).
static __device__ __forceinline__ void tdm_load_v_tile(const __bf16* gsrc, unsigned lds_byte_off) {
  unsigned long long ga = (unsigned long long)(uintptr_t)gsrc;
  v4u g0;
  g0[0] = 1u;                                            // count=1, user descriptor
  g0[1] = lds_byte_off;                                  // lds_addr
  g0[2] = (unsigned)(ga & 0xFFFFFFFFu);                  // global_addr[31:0]
  g0[3] = (unsigned)((ga >> 32) & 0x1FFFFFFu) | (2u << 30);  // global_addr[56:32] | type=2
  v8i g1;
  g1[0] = (int)((1u << 16)      // data_size = 2 bytes
              | (1u << 20)      // pad_enable
              | (5u << 22)      // pad_interval: 64 DWORDs (256B)
              | (3u << 25));    // pad_amount: 4 DWORDs (16B)
  g1[1] = (int)(128u << 16);    // tensor_dim0 = 128 (d)
  g1[2] = (int)(64u << 16);     // tensor_dim1 = 64 (keys)
  g1[3] = (int)(128u << 16);    // tile_dim0 = 128
  g1[4] = 64;                   // tile_dim1 = 64
  g1[5] = 128;                  // tensor_dim0_stride = HD
  g1[6] = 0;
  g1[7] = 0;
  v4i gz = {};
#if __clang_major__ >= 23
  v8i gz8 = {};
  __builtin_amdgcn_tensor_load_to_lds(g0, g1, gz, gz, gz8, 0);
#else
  __builtin_amdgcn_tensor_load_to_lds(g0, g1, gz, gz, 0);
#endif
}

// MODE: 0 = plain f32 out (O projection), 1 = Q (rmsnorm+rope), 2 = K (rmsnorm+rope), 3 = V (rmsnorm)
template<int MODE>
__global__ __launch_bounds__(256) void gemm_kernel(
    const float* __restrict__ A, const float* __restrict__ Bw,
    const float* __restrict__ normw, const long long* __restrict__ pos,
    float* __restrict__ outf, __bf16* __restrict__ outbf,
    int N, int nheads)
{
  __shared__ union {
    struct { __bf16 a[128 * 32]; __bf16 b[32 * 128]; } s;
    float c[128 * 129];
  } lds;

  const int tid  = threadIdx.x;
  const int lane = tid & 31;
  const int wid  = tid >> 5;
  const int n0   = blockIdx.x * 128;
  const int m0   = blockIdx.y * 128;
  const int wm   = (wid >> 2) * 64;   // 0 or 64
  const int wn   = (wid & 3) * 32;    // 0,32,64,96

  v8f acc[4][2] = {};

  for (int k0 = 0; k0 < HIDDEN; k0 += 32) {
    // stage A tile 128x32 (f32 -> bf16, packed b64 stores)
    #pragma unroll
    for (int it = 0; it < 4; ++it) {
      int idx = tid + it * 256;
      int row = idx >> 3, c4 = (idx & 7) << 2;
      v4f v = *(const v4f*)(A + (size_t)(m0 + row) * HIDDEN + k0 + c4);
      unsigned long long p = (unsigned long long)pack_bf16(v.x, v.y)
                           | ((unsigned long long)pack_bf16(v.z, v.w) << 32);
      *(unsigned long long*)(&lds.s.a[row * 32 + c4]) = p;
    }
    // stage B tile 32x128
    #pragma unroll
    for (int it = 0; it < 4; ++it) {
      int idx = tid + it * 256;
      int row = idx >> 5, c4 = (idx & 31) << 2;
      v4f v = *(const v4f*)(Bw + (size_t)(k0 + row) * N + n0 + c4);
      unsigned long long p = (unsigned long long)pack_bf16(v.x, v.y)
                           | ((unsigned long long)pack_bf16(v.z, v.w) << 32);
      *(unsigned long long*)(&lds.s.b[row * 128 + c4]) = p;
    }
    // prefetch next K-step tiles into cache (global_prefetch_b8)
    if (k0 + 32 < HIDDEN) {
      __builtin_prefetch(A + (size_t)(m0 + (tid >> 1)) * HIDDEN + k0 + 32 + (tid & 1) * 16, 0, 0);
      __builtin_prefetch(Bw + (size_t)(k0 + 32 + (tid >> 3)) * N + n0 + (tid & 7) * 16, 0, 0);
    }
    __syncthreads();

    v16bf bfr[2], afr[4];
    #pragma unroll
    for (int j = 0; j < 2; ++j) bfr[j] = ldsB(&lds.s.b[wn + j * 16], 128, lane);
    #pragma unroll
    for (int i = 0; i < 4; ++i) afr[i] = ldsA(&lds.s.a[(wm + i * 16) * 32], 32, lane);
    #pragma unroll
    for (int i = 0; i < 4; ++i)
      #pragma unroll
      for (int j = 0; j < 2; ++j) acc[i][j] = wmma_bf16(afr[i], bfr[j], acc[i][j]);
    __syncthreads();
  }

  // C fragments -> LDS (f32, stride 129 to dodge bank conflicts)
  #pragma unroll
  for (int i = 0; i < 4; ++i)
    #pragma unroll
    for (int j = 0; j < 2; ++j)
      #pragma unroll
      for (int r = 0; r < 8; ++r)
        lds.c[(wm + i * 16 + r + ((lane >> 4) << 3)) * 129 + wn + j * 16 + (lane & 15)] = acc[i][j][r];
  __syncthreads();

  if constexpr (MODE == 0) {
    for (int idx = tid; idx < 128 * 32; idx += 256) {
      int row = idx >> 5, c4 = (idx & 31) << 2;
      const float* s = &lds.c[row * 129 + c4];
      v4f v; v.x = s[0]; v.y = s[1]; v.z = s[2]; v.w = s[3];
      *(v4f*)(outf + (size_t)(m0 + row) * N + n0 + c4) = v;
    }
  } else {
    // one 128-wide head per N-tile; 2 threads per row (64 dims each = one RoPE sub-block)
    int row = tid >> 1;
    int half = tid & 1;
    int token = m0 + row;
    int b = token / SEQ, s = token % SEQ;
    int head = blockIdx.x;
    float v[64];
    float ssq = 0.f;
    #pragma unroll
    for (int j = 0; j < 64; ++j) { v[j] = lds.c[row * 129 + half * 64 + j]; ssq += v[j] * v[j]; }
    ssq += __shfl_xor(ssq, 1, 32);
    float rms = rsqrtf(ssq * (1.f / 128.f) + 1e-6f);
    __bf16* dst = outbf + ((size_t)(b * nheads + head) * SEQ + s) * HD + half * 64;
    if constexpr (MODE == 3) {
      #pragma unroll
      for (int j = 0; j < 64; ++j) dst[j] = (__bf16)(v[j] * rms);
    } else {
      float p0 = (float)pos[(size_t)token * 2 + 0];
      float p1 = (float)pos[(size_t)token * 2 + 1];
      #pragma unroll
      for (int j = 0; j < 32; ++j) {
        float w0 = normw[half * 64 + j];
        float w1 = normw[half * 64 + 32 + j];
        float a = v[j] * rms * w0;
        float bb = v[j + 32] * rms * w1;
        float f = __powf(10000.f, -(float)j * (1.f / 32.f));
        float a0 = p0 * f, a1 = p1 * f;
        dst[j]      = (__bf16)(a * __cosf(a0) - bb * __sinf(a0));
        dst[j + 32] = (__bf16)(bb * __cosf(a1) + a * __sinf(a1));
      }
    }
  }
}

__global__ __launch_bounds__(128) void flash_kernel(
    const __bf16* __restrict__ Qbf, const __bf16* __restrict__ Kbf,
    const __bf16* __restrict__ Vbf, const long long* __restrict__ pos,
    float* __restrict__ Yf)
{
  constexpr int KT_STRIDE = 72;   // multiples of 8 elems ; Kt frag reads (ldsB) are strided 16-bit loads
  constexpr int V_STRIDE  = 136;  // 128 + 8 elems: matches TDM pad (16B per 256B)
  constexpr int P_STRIDE  = 72;
  __shared__ __attribute__((aligned(16))) __bf16 Kt[128 * KT_STRIDE];     // [d][key] transposed
  __shared__ __attribute__((aligned(16))) __bf16 Vs[2][64 * V_STRIDE];    // [key][d], double buffered (TDM)
  __shared__ __attribute__((aligned(16))) __bf16 Ps[4][16 * P_STRIDE];    // per-wave P tile
  __shared__ float maskS[64];

  const int tid  = threadIdx.x;
  const int lane = tid & 31;
  const int wid  = tid >> 5;
  const int b    = blockIdx.z;
  const int h    = blockIdx.y;
  const int kv   = h / (NQ / NKV);
  const int q0   = blockIdx.x * 64 + wid * 16;

  const __bf16* kbase = Kbf + (size_t)(b * NKV + kv) * SEQ * HD;
  const __bf16* vbase = Vbf + (size_t)(b * NKV + kv) * SEQ * HD;

  // kick off the first V tile DMA (one wave owns TENSORcnt)
  if (wid == 0) tdm_load_v_tile(vbase, (unsigned)(uintptr_t)(&Vs[0][0]));

  // Q fragments straight from global (A layout), 8 chunks of K=32 over d
  v16bf qf[8];
  const __bf16* qbase = Qbf + ((size_t)(b * NQ + h) * SEQ + q0 + (lane & 15)) * HD;
  #pragma unroll
  for (int c = 0; c < 8; ++c) {
    U16 u;
    u.h[0] = *(const v8bf*)(qbase + c * 32 + ((lane >> 4) << 3));
    u.h[1] = *(const v8bf*)(qbase + c * 32 + 16 + ((lane >> 4) << 3));
    qf[c] = u.v;
  }

  v8f o[8] = {};
  float mrow[8], lrow[8];
  #pragma unroll
  for (int r = 0; r < 8; ++r) { mrow[r] = -INFINITY; lrow[r] = 0.f; }

  for (int t = 0; t < SEQ / 64; ++t) {
    int key0 = t * 64;
    // stage K transposed (manual: TDM cannot transpose)
    #pragma unroll
    for (int it = 0; it < 8; ++it) {
      int idx = tid + it * 128;
      int key = idx >> 4, c = (idx & 15) << 3;
      v8bf kvv = *(const v8bf*)(kbase + (size_t)(key0 + key) * HD + c);
      #pragma unroll
      for (int j = 0; j < 8; ++j) Kt[(c + j) * KT_STRIDE + key] = kvv[j];
    }
    if (tid < 64) {
      long long p0v = pos[((size_t)b * SEQ + key0 + tid) * 2 + 0];
      long long p1v = pos[((size_t)b * SEQ + key0 + tid) * 2 + 1];
      maskS[tid] = (p0v == -1 && p1v == -1) ? -3.0e38f : 0.f;
    }
    if (t + 1 < SEQ / 64)
      __builtin_prefetch(kbase + (size_t)(key0 + 64 + (tid >> 1)) * HD + (tid & 1) * 64, 0, 0);
    if (wid == 0) __builtin_amdgcn_s_wait_tensorcnt(0);   // V tile t DMA done
    __syncthreads();

    const __bf16* vs = &Vs[t & 1][0];

    // S = Q K^T : 4 key sub-tiles of 16, accumulate over 4 d-chunks of 32
    v8f sfr[4];
    #pragma unroll
    for (int f = 0; f < 4; ++f) {
      v16bf kb[4];
      #pragma unroll
      for (int c = 0; c < 4; ++c) kb[c] = ldsB(&Kt[(c * 32) * KT_STRIDE + f * 16], KT_STRIDE, lane);
      v8f acc = {};
      #pragma unroll
      for (int c = 0; c < 4; ++c) acc = wmma_bf16(qf[c], kb[c], acc);
      float mk = maskS[f * 16 + (lane & 15)];
      #pragma unroll
      for (int r = 0; r < 8; ++r) acc[r] += mk;
      sfr[f] = acc;
    }

    // online softmax (row stats redundant across each 16-lane group)
    float mnew[8], scale[8];
    #pragma unroll
    for (int r = 0; r < 8; ++r) {
      float mx = fmaxf(fmaxf(sfr[0][r], sfr[1][r]), fmaxf(sfr[2][r], sfr[3][r]));
      mx = fmaxf(mx, __shfl_xor(mx, 1, 32));
      mx = fmaxf(mx, __shfl_xor(mx, 2, 32));
      mx = fmaxf(mx, __shfl_xor(mx, 4, 32));
      mx = fmaxf(mx, __shfl_xor(mx, 8, 32));
      float mn = fmaxf(mrow[r], mx);
      scale[r] = (mrow[r] == -INFINITY) ? 0.f : __expf(mrow[r] - mn);
      mnew[r] = mn;
    }
    float rs[8] = {};
    #pragma unroll
    for (int f = 0; f < 4; ++f)
      #pragma unroll
      for (int r = 0; r < 8; ++r) {
        float p = __expf(sfr[f][r] - mnew[r]);
        sfr[f][r] = p;
        rs[r] += p;
      }
    #pragma unroll
    for (int r = 0; r < 8; ++r) {
      rs[r] += __shfl_xor(rs[r], 1, 32);
      rs[r] += __shfl_xor(rs[r], 2, 32);
      rs[r] += __shfl_xor(rs[r], 4, 32);
      rs[r] += __shfl_xor(rs[r], 8, 32);
      lrow[r] = lrow[r] * scale[r] + rs[r];
      mrow[r] = mnew[r];
    }
    #pragma unroll
    for (int n = 0; n < 8; ++n)
      #pragma unroll
      for (int r = 0; r < 8; ++r) o[n][r] *= scale[r];

    // P: C layout -> per-wave LDS -> A layout
    __bf16* pw = &Ps[wid][0];
    #pragma unroll
    for (int f = 0; f < 4; ++f)
      #pragma unroll
      for (int r = 0; r < 8; ++r)
        pw[(r + ((lane >> 4) << 3)) * P_STRIDE + f * 16 + (lane & 15)] = (__bf16)sfr[f][r];
    __builtin_amdgcn_wave_barrier();
    asm volatile("s_wait_dscnt 0" ::: "memory");
    __builtin_amdgcn_wave_barrier();
    v16bf pf[2];
    #pragma unroll
    for (int kc = 0; kc < 2; ++kc) pf[kc] = ldsA(&pw[kc * 32], P_STRIDE, lane);

    // O += P V
    #pragma unroll
    for (int n = 0; n < 8; ++n) {
      v16bf vb[2];
      #pragma unroll
      for (int kc = 0; kc < 2; ++kc) vb[kc] = ldsB(&vs[(kc * 32) * V_STRIDE + n * 16], V_STRIDE, lane);
      #pragma unroll
      for (int kc = 0; kc < 2; ++kc) o[n] = wmma_bf16(pf[kc], vb[kc], o[n]);
    }

    // overlap: launch next V tile DMA into the other buffer (safe: every wave
    // has passed this iteration's mid barrier, so tile t-1 reads are done)
    if (wid == 0 && t + 1 < SEQ / 64)
      tdm_load_v_tile(vbase + (size_t)(key0 + 64) * HD, (unsigned)(uintptr_t)(&Vs[(t + 1) & 1][0]));
    __syncthreads();
  }

  // finalize: divide by l, store f32 Y[token][h*128+d] for the O projection
  #pragma unroll
  for (int r = 0; r < 8; ++r) {
    float inv = (lrow[r] > 0.f) ? 1.f / lrow[r] : 0.f;
    int srow = q0 + r + ((lane >> 4) << 3);
    float* dst = Yf + ((size_t)b * SEQ + srow) * (NQ * HD) + h * HD + (lane & 15);
    #pragma unroll
    for (int n = 0; n < 8; ++n) dst[n * 16] = o[n][r] * inv;
  }
}

extern "C" void kernel_launch(void* const* d_in, const int* in_sizes, int n_in,
                              void* d_out, int out_size, void* d_ws, size_t ws_size,
                              hipStream_t stream) {
  const float* x        = (const float*)d_in[0];
  const long long* pos  = (const long long*)d_in[1];
  const float* q_w      = (const float*)d_in[2];
  const float* k_w      = (const float*)d_in[3];
  const float* v_w      = (const float*)d_in[4];
  const float* o_w      = (const float*)d_in[5];
  const float* q_nw     = (const float*)d_in[6];
  const float* k_nw     = (const float*)d_in[7];

  char* ws = (char*)d_ws;
  size_t szQ = (size_t)BATCH * NQ * SEQ * HD * 2;   // bf16
  size_t szK = (size_t)BATCH * NKV * SEQ * HD * 2;  // bf16
  __bf16* Qbf = (__bf16*)(ws);
  __bf16* Kbf = (__bf16*)(ws + szQ);
  __bf16* Vbf = (__bf16*)(ws + szQ + szK);
  float*  Yf  = (float*)(ws + szQ + 2 * szK);

  dim3 blk(256);
  gemm_kernel<1><<<dim3(NQ, 32),  blk, 0, stream>>>(x, q_w, q_nw, pos, nullptr, Qbf, NQ * HD, NQ);
  gemm_kernel<2><<<dim3(NKV, 32), blk, 0, stream>>>(x, k_w, k_nw, pos, nullptr, Kbf, NKV * HD, NKV);
  gemm_kernel<3><<<dim3(NKV, 32), blk, 0, stream>>>(x, v_w, nullptr, pos, nullptr, Vbf, NKV * HD, NKV);
  flash_kernel<<<dim3(SEQ / 64, NQ, BATCH), dim3(128), 0, stream>>>(Qbf, Kbf, Vbf, pos, Yf);
  gemm_kernel<0><<<dim3(NQ, 32), blk, 0, stream>>>(Yf, o_w, nullptr, nullptr, (float*)d_out, nullptr, HIDDEN, 0);
}
